// DeepseekV2MLPMoE_13907104105108
// MI455X (gfx1250) — matrix operand order repaired and checked by reading, not compiled, not run on hardware
//
#include <hip/hip_runtime.h>
#include <hip/hip_bf16.h>

#define NUM_TOKENS 16384
#define HIDDEN 2048
#define INTER 1408
#define N_EXPERTS 32

#define TM 128          // block M tile (4 waves x 32 rows)
#define TN 64           // block N tile
#define KB 32           // K step (one bf16 WMMA K)
#define THREADS 128
#define LDP 20          // B tile LDS row stride in dwords (16 data + 4 pad; 80B, 16B-aligned)
#define A1S 36          // kernel1 A tile LDS row stride in dwords (32 f32 + pad; 144B, 16B-aligned)
#define A2S 20          // kernel2 A tile LDS row stride in dwords (16 bf16x2 + pad; 80B, 16B-aligned)

typedef __attribute__((ext_vector_type(16))) __bf16 v16bf;
typedef __attribute__((ext_vector_type(8)))  float  v8f;

union Frag { v16bf v; unsigned u[8]; uint4 q[2]; };

__device__ __forceinline__ unsigned pack_bf16(float a, float b) {
    __hip_bfloat162 h = __float22bfloat162_rn(float2{a, b});
    unsigned r;
    __builtin_memcpy(&r, &h, sizeof(r));
    return r;
}

__device__ __forceinline__ int expert_of_row(const int* __restrict__ gsz, int m0) {
    int e = 0, start = 0;
#pragma unroll
    for (int i = 0; i < N_EXPERTS; i++) {
        int g = gsz[i];
        if (m0 >= start + g) { start += g; e++; }
    }
    return e;
}

// async copy of 16 bytes/lane from global to LDS (CDNA5 GLOBAL_LOAD_ASYNC_TO_LDS)
__device__ __forceinline__ void async_b128(unsigned lds_off, const void* gaddr) {
    asm volatile("global_load_async_to_lds_b128 %0, %1, off"
                 :: "v"(lds_off), "v"((unsigned long long)gaddr) : "memory");
}
__device__ __forceinline__ void wait_async0() {
    asm volatile("s_wait_asynccnt 0x0" ::: "memory");
}

// ---------------------------------------------------------------------------
// Kernel 1: act = silu(X @ Wg[e]) * (X @ Wu[e]),  act stored as bf16 [M, INTER]
// ---------------------------------------------------------------------------
__global__ __launch_bounds__(THREADS) void moe_gateup_kernel(
    const float* __restrict__ X,
    const float* __restrict__ Wg,
    const float* __restrict__ Wu,
    const int*   __restrict__ gsz,
    __hip_bfloat16* __restrict__ act)
{
    __shared__ __attribute__((aligned(16))) float    Afp[2][TM * A1S];
    __shared__ __attribute__((aligned(16))) unsigned Bg[2][TN * LDP];
    __shared__ __attribute__((aligned(16))) unsigned Bu[2][TN * LDP];

    const int m0 = blockIdx.x * TM;
    const int n0 = blockIdx.y * TN;

    const int e = expert_of_row(gsz, m0);
    const float* wg = Wg + (size_t)e * HIDDEN * INTER;
    const float* wu = Wu + (size_t)e * HIDDEN * INTER;

    const int t    = threadIdx.x;
    const int lane = t & 31;
    const int wave = t >> 5;
    const int l15  = lane & 15;
    const int hi   = lane >> 4;

    v8f accg[2][4], accu[2][4];
#pragma unroll
    for (int ms = 0; ms < 2; ms++)
#pragma unroll
        for (int ns = 0; ns < 4; ns++) { accg[ms][ns] = (v8f){}; accu[ms][ns] = (v8f){}; }

#define STAGE_A1(buf, kk)                                                          \
    {                                                                              \
        _Pragma("unroll")                                                          \
        for (int c = 0; c < 8; c++) {                                              \
            int bi  = (c * THREADS + t) * 16;   /* byte index in 16KB tile */      \
            int row = bi >> 7;                  /* 128B of fp32 per row */         \
            int bo  = bi & 127;                                                    \
            unsigned lds = (unsigned)(size_t)((char*)&Afp[buf][0] + row * (A1S*4) + bo); \
            const char* ga = (const char*)(X + (size_t)(m0 + row) * HIDDEN + (kk)) + bo; \
            async_b128(lds, ga);                                                   \
        }                                                                          \
    }

#define STAGE_B1(buf, kk)                                                          \
    {                                                                              \
        _Pragma("unroll")                                                          \
        for (int i = 0; i < 2; i++) {                                              \
            int qp = i * THREADS + t;           /* 256 float4 groups */            \
            int n4 = (qp & 15) * 4;                                                \
            int cp = qp >> 4;                   /* k-pair 0..15 */                 \
            size_t off = (size_t)((kk) + 2 * cp) * INTER + n0 + n4;                \
            float4 g0 = *(const float4*)(wg + off);                                \
            float4 g1 = *(const float4*)(wg + off + INTER);                        \
            float4 u0 = *(const float4*)(wu + off);                                \
            float4 u1 = *(const float4*)(wu + off + INTER);                        \
            const float* pg0 = (const float*)&g0; const float* pg1 = (const float*)&g1; \
            const float* pu0 = (const float*)&u0; const float* pu1 = (const float*)&u1; \
            _Pragma("unroll")                                                      \
            for (int j = 0; j < 4; j++) {                                          \
                Bg[buf][(n4 + j) * LDP + cp] = pack_bf16(pg0[j], pg1[j]);          \
                Bu[buf][(n4 + j) * LDP + cp] = pack_bf16(pu0[j], pu1[j]);          \
            }                                                                      \
        }                                                                          \
    }

#define COMPUTE1(buf)                                                              \
    {                                                                              \
        Frag a[2];                                                                 \
        _Pragma("unroll")                                                          \
        for (int ms = 0; ms < 2; ms++) {                                           \
            /* per-lane base: row mrow, fp32 k offset hi*8; all chunk offsets constant */ \
            const float* abase = &Afp[buf][(wave * 32 + ms * 16 + l15) * A1S + (hi << 3)]; \
            float4 f0 = *(const float4*)(abase + 0);   /* k: 0..3  (+hi*8) */      \
            float4 f1 = *(const float4*)(abase + 4);   /* k: 4..7  */              \
            float4 f2 = *(const float4*)(abase + 16);  /* k: 16..19 */             \
            float4 f3 = *(const float4*)(abase + 20);  /* k: 20..23 */             \
            a[ms].u[0] = pack_bf16(f0.x, f0.y); a[ms].u[1] = pack_bf16(f0.z, f0.w);\
            a[ms].u[2] = pack_bf16(f1.x, f1.y); a[ms].u[3] = pack_bf16(f1.z, f1.w);\
            a[ms].u[4] = pack_bf16(f2.x, f2.y); a[ms].u[5] = pack_bf16(f2.z, f2.w);\
            a[ms].u[6] = pack_bf16(f3.x, f3.y); a[ms].u[7] = pack_bf16(f3.z, f3.w);\
        }                                                                          \
        const unsigned* bgb = &Bg[buf][l15 * LDP + (hi << 2)];                     \
        const unsigned* bub = &Bu[buf][l15 * LDP + (hi << 2)];                     \
        _Pragma("unroll")                                                          \
        for (int ns = 0; ns < 4; ns++) {                                           \
            Frag bgf, buf_;                                                        \
            bgf.q[0]  = *(const uint4*)(bgb + ns * 16 * LDP);                      \
            bgf.q[1]  = *(const uint4*)(bgb + ns * 16 * LDP + 8);                  \
            buf_.q[0] = *(const uint4*)(bub + ns * 16 * LDP);                      \
            buf_.q[1] = *(const uint4*)(bub + ns * 16 * LDP + 8);                  \
            _Pragma("unroll")                                                      \
            for (int ms = 0; ms < 2; ms++) {                                       \
                accg[ms][ns] = __builtin_amdgcn_wmma_f32_16x16x32_bf16(            \
                    false, a[ms].v, false, bgf.v,  (short)0, accg[ms][ns], false, false); \
                accu[ms][ns] = __builtin_amdgcn_wmma_f32_16x16x32_bf16(            \
                    false, a[ms].v, false, buf_.v, (short)0, accu[ms][ns], false, false); \
            }                                                                      \
        }                                                                          \
    }

    STAGE_A1(0, 0);
    STAGE_B1(0, 0);
    wait_async0();
    __syncthreads();

    const int KT = HIDDEN / KB;   // 64
    for (int kt = 0; kt < KT; kt++) {
        const int cur = kt & 1;
        if (kt + 1 < KT) {
            STAGE_A1(cur ^ 1, (kt + 1) * KB);
            STAGE_B1(cur ^ 1, (kt + 1) * KB);
        }
        COMPUTE1(cur);
        wait_async0();
        __syncthreads();
    }

    // epilogue: silu(g)*u -> bf16; C layout: m = r + 8*hi, n = l15
#pragma unroll
    for (int ms = 0; ms < 2; ms++) {
        const int mbase = m0 + wave * 32 + ms * 16 + (hi << 3);
        const int nbase = n0 + l15;
#pragma unroll
        for (int ns = 0; ns < 4; ns++) {
#pragma unroll
            for (int r = 0; r < 8; r++) {
                float g = accg[ms][ns][r];
                float u = accu[ms][ns][r];
                float s = g * __builtin_amdgcn_rcpf(1.0f + __expf(-g));
                act[(size_t)(mbase + r) * INTER + nbase + ns * 16] =
                    __float2bfloat16(s * u);
            }
        }
    }
}

// ---------------------------------------------------------------------------
// Kernel 2: out = act @ Wd[e],  out fp32 [M, HIDDEN]
// ---------------------------------------------------------------------------
__global__ __launch_bounds__(THREADS) void moe_down_kernel(
    const __hip_bfloat16* __restrict__ act,
    const float* __restrict__ Wd,
    const int*   __restrict__ gsz,
    float* __restrict__ out)
{
    __shared__ __attribute__((aligned(16))) unsigned Abf[2][TM * A2S];
    __shared__ __attribute__((aligned(16))) unsigned Bd[2][TN * LDP];

    const int m0 = blockIdx.x * TM;
    const int n0 = blockIdx.y * TN;

    const int e = expert_of_row(gsz, m0);
    const float* wd = Wd + (size_t)e * INTER * HIDDEN;
    const char*  actb = (const char*)act;

    const int t    = threadIdx.x;
    const int lane = t & 31;
    const int wave = t >> 5;
    const int l15  = lane & 15;
    const int hi   = lane >> 4;

    v8f acc[2][4];
#pragma unroll
    for (int ms = 0; ms < 2; ms++)
#pragma unroll
        for (int ns = 0; ns < 4; ns++) acc[ms][ns] = (v8f){};

#define STAGE_A2(buf, kk)                                                          \
    {                                                                              \
        _Pragma("unroll")                                                          \
        for (int c = 0; c < 4; c++) {                                              \
            int bi  = (c * THREADS + t) * 16;   /* byte index in 8KB tile */       \
            int row = bi >> 6;                  /* 64B of bf16 per row */          \
            int bo  = bi & 63;                                                     \
            unsigned lds = (unsigned)(size_t)((char*)&Abf[buf][0] + row * (A2S*4) + bo); \
            const char* ga = actb + (size_t)(m0 + row) * (INTER * 2) + (kk) * 2 + bo; \
            async_b128(lds, ga);                                                   \
        }                                                                          \
    }

#define STAGE_B2(buf, kk)                                                          \
    {                                                                              \
        _Pragma("unroll")                                                          \
        for (int i = 0; i < 2; i++) {                                              \
            int qp = i * THREADS + t;                                              \
            int n4 = (qp & 15) * 4;                                                \
            int cp = qp >> 4;                                                      \
            size_t off = (size_t)((kk) + 2 * cp) * HIDDEN + n0 + n4;               \
            float4 d0 = *(const float4*)(wd + off);                                \
            float4 d1 = *(const float4*)(wd + off + HIDDEN);                       \
            const float* p0 = (const float*)&d0; const float* p1 = (const float*)&d1; \
            _Pragma("unroll")                                                      \
            for (int j = 0; j < 4; j++)                                            \
                Bd[buf][(n4 + j) * LDP + cp] = pack_bf16(p0[j], p1[j]);            \
        }                                                                          \
    }

#define COMPUTE2(buf)                                                              \
    {                                                                              \
        Frag a[2];                                                                 \
        _Pragma("unroll")                                                          \
        for (int ms = 0; ms < 2; ms++) {                                           \
            const unsigned* abase = &Abf[buf][(wave * 32 + ms * 16 + l15) * A2S + (hi << 2)]; \
            a[ms].q[0] = *(const uint4*)(abase + 0);   /* k-pairs 0..3 (+hi*4) */  \
            a[ms].q[1] = *(const uint4*)(abase + 8);   /* k-pairs 8..11 */         \
        }                                                                          \
        const unsigned* bdb = &Bd[buf][l15 * LDP + (hi << 2)];                     \
        _Pragma("unroll")                                                          \
        for (int ns = 0; ns < 4; ns++) {                                           \
            Frag b;                                                                \
            b.q[0] = *(const uint4*)(bdb + ns * 16 * LDP);                         \
            b.q[1] = *(const uint4*)(bdb + ns * 16 * LDP + 8);                     \
            _Pragma("unroll")                                                      \
            for (int ms = 0; ms < 2; ms++)                                         \
                acc[ms][ns] = __builtin_amdgcn_wmma_f32_16x16x32_bf16(             \
                    false, a[ms].v, false, b.v, (short)0, acc[ms][ns], false, false); \
        }                                                                          \
    }

    STAGE_A2(0, 0);
    STAGE_B2(0, 0);
    wait_async0();
    __syncthreads();

    const int KT = INTER / KB;   // 44
    for (int kt = 0; kt < KT; kt++) {
        const int cur = kt & 1;
        if (kt + 1 < KT) {
            STAGE_A2(cur ^ 1, (kt + 1) * KB);
            STAGE_B2(cur ^ 1, (kt + 1) * KB);
        }
        COMPUTE2(cur);
        wait_async0();
        __syncthreads();
    }

#pragma unroll
    for (int ms = 0; ms < 2; ms++) {
        const int mbase = m0 + wave * 32 + ms * 16 + (hi << 3);
        const int nbase = n0 + l15;
#pragma unroll
        for (int ns = 0; ns < 4; ns++) {
#pragma unroll
            for (int r = 0; r < 8; r++) {
                out[(size_t)(mbase + r) * HIDDEN + nbase + ns * 16] = acc[ms][ns][r];
            }
        }
    }
}

extern "C" void kernel_launch(void* const* d_in, const int* in_sizes, int n_in,
                              void* d_out, int out_size, void* d_ws, size_t ws_size,
                              hipStream_t stream) {
    const float* X  = (const float*)d_in[0];
    const float* Wg = (const float*)d_in[1];
    const float* Wu = (const float*)d_in[2];
    const float* Wd = (const float*)d_in[3];
    const int*   gs = (const int*)d_in[4];
    float* out = (float*)d_out;
    __hip_bfloat16* act = (__hip_bfloat16*)d_ws; // 16384*1408*2 ~= 46 MB scratch

    dim3 block(THREADS);
    dim3 grid1(NUM_TOKENS / TM, INTER / TN);   // 128 x 22
    moe_gateup_kernel<<<grid1, block, 0, stream>>>(X, Wg, Wu, gs, act);

    dim3 grid2(NUM_TOKENS / TM, HIDDEN / TN);  // 128 x 32
    moe_down_kernel<<<grid2, block, 0, stream>>>(act, Wd, gs, out);
}
